// DynamicConv2d_15023795601737
// MI455X (gfx1250) — matrix operand order repaired and checked by reading, not compiled, not run on hardware
//
#include <hip/hip_runtime.h>
#include <hip/hip_bf16.h>
#include <math.h>

typedef __attribute__((ext_vector_type(8)))  _Float16 h8;
typedef __attribute__((ext_vector_type(16))) _Float16 h16;
typedef __attribute__((ext_vector_type(8)))  float    v8f;
typedef __attribute__((ext_vector_type(4)))  float    vf4;

#define B_   32
#define C_   128
#define H_   56
#define W_   56
#define HW_  (H_*W_)          // 3136
#define COUT 128
#define KK   4
#define CRED 32
#define KDIM (9*C_)           // 1152, K ordered (kh,kw,c)
#define TEMP 30.0f
#define NX   (B_*C_*HW_)      // 12,845,056

// ---------------------------------------------------------------------------
// 1) Global average pool: pooled[b][c] = mean_{hw} x[b][c][hw]
__global__ __launch_bounds__(128) void pool_kernel(const float* __restrict__ x,
                                                   float* __restrict__ pooled) {
    const int c = blockIdx.x, b = blockIdx.y, tid = threadIdx.x;
    const float* xp = x + ((size_t)(b * C_ + c)) * HW_;
    float s = 0.f;
    for (int i = tid; i < HW_; i += 128) s += xp[i];
    __shared__ float red[128];
    red[tid] = s;
    __syncthreads();
    for (int off = 64; off > 0; off >>= 1) {
        if (tid < off) red[tid] += red[tid + off];
        __syncthreads();
    }
    if (tid == 0) pooled[b * C_ + c] = red[0] * (1.0f / (float)HW_);
}

// ---------------------------------------------------------------------------
// 2) Attention MLP + softmax(logits / T). 1 block, 32 threads (1 per batch).
__global__ __launch_bounds__(32) void attn_kernel(const float* __restrict__ pooled,
                                                  const float* __restrict__ aw1,
                                                  const float* __restrict__ aw2,
                                                  float* __restrict__ attn) {
    const int b = threadIdx.x;
    const float* p = pooled + b * C_;
    float lg[KK] = {0.f, 0.f, 0.f, 0.f};
    for (int r = 0; r < CRED; ++r) {
        float s = 0.f;
        const float* w = aw1 + r * C_;
        for (int c = 0; c < C_; ++c) s += p[c] * w[c];
        const float h1 = fmaxf(s, 0.f);
        #pragma unroll
        for (int k = 0; k < KK; ++k) lg[k] += h1 * aw2[k * CRED + r];
    }
    float m = -1e30f;
    #pragma unroll
    for (int k = 0; k < KK; ++k) { lg[k] *= (1.0f / TEMP); m = fmaxf(m, lg[k]); }
    float sum = 0.f, e[KK];
    #pragma unroll
    for (int k = 0; k < KK; ++k) { e[k] = __expf(lg[k] - m); sum += e[k]; }
    const float inv = 1.0f / sum;
    #pragma unroll
    for (int k = 0; k < KK; ++k) attn[b * KK + k] = e[k] * inv;
}

// ---------------------------------------------------------------------------
// 3) Mix weights: weff[b][o][p*128+c] = f16( sum_k attn[b][k]*W[k][o][c][p] )
__global__ __launch_bounds__(256) void weff_kernel(const float* __restrict__ W,
                                                   const float* __restrict__ bvec,
                                                   const float* __restrict__ attn,
                                                   _Float16* __restrict__ weff,
                                                   float* __restrict__ beff) {
    const int o = blockIdx.x, b = blockIdx.y, tid = threadIdx.x;
    const float a0 = attn[b * KK + 0], a1 = attn[b * KK + 1];
    const float a2 = attn[b * KK + 2], a3 = attn[b * KK + 3];
    const size_t kstride = (size_t)COUT * C_ * 9;
    const float* w0 = W + 0 * kstride + (size_t)o * (C_ * 9);
    const float* w1 = W + 1 * kstride + (size_t)o * (C_ * 9);
    const float* w2 = W + 2 * kstride + (size_t)o * (C_ * 9);
    const float* w3 = W + 3 * kstride + (size_t)o * (C_ * 9);
    _Float16* dst = weff + ((size_t)(b * COUT + o)) * KDIM;
    for (int e = tid; e < KDIM; e += 256) {
        const int p = e >> 7;        // tap
        const int c = e & 127;       // channel
        const int src = c * 9 + p;   // W inner layout [c][kh][kw]
        const float v = a0 * w0[src] + a1 * w1[src] + a2 * w2[src] + a3 * w3[src];
        dst[e] = (_Float16)v;
    }
    if (tid == 0)
        beff[b * COUT + o] = a0 * bvec[0 * COUT + o] + a1 * bvec[1 * COUT + o] +
                             a2 * bvec[2 * COUT + o] + a3 * bvec[3 * COUT + o];
}

// ---------------------------------------------------------------------------
// 4) NCHW f32 -> NHWC f16 tiled transpose (32c x 64n per block via LDS).
// grid (49, 4, 32), 256 threads
__global__ __launch_bounds__(256) void nhwc_kernel(const float* __restrict__ x,
                                                   _Float16* __restrict__ xh) {
    const int b = blockIdx.z, c0 = blockIdx.y * 32, n0 = blockIdx.x * 64;
    __shared__ _Float16 t[32][66];   // +2 pad: conflict-free transpose read
    const int tid = threadIdx.x;
    const float* xp = x + ((size_t)(b * C_ + c0)) * HW_ + n0;
    #pragma unroll
    for (int i = 0; i < 8; ++i) {
        const int idx = tid + i * 256;        // 0..2047
        const int cc = idx >> 6, nn = idx & 63;
        t[cc][nn] = (_Float16)xp[(size_t)cc * HW_ + nn];   // coalesced along n
    }
    __syncthreads();
    _Float16* op = xh + ((size_t)b * HW_ + n0) * C_ + c0;
    #pragma unroll
    for (int i = 0; i < 8; ++i) {
        const int idx = tid + i * 256;
        const int nn = idx >> 5, cc = idx & 31;
        op[(size_t)nn * C_ + cc] = t[cc][nn];              // coalesced along c
    }
}

// ---------------------------------------------------------------------------
// 5) Implicit-GEMM conv, barrier-free main loop.
//    Block = 256 thr (8 waves), tile M=128 (Cout) x N=64. Wave w: 16 M-rows,
//    4 accumulators of 16 N. B fragments load straight from NHWC xh:
//    16 consecutive channels at one pixel = 32 contiguous bytes per lane.
__device__ __forceinline__ h16 load_b_frag(const _Float16* p, bool ok) {
    h16 z = {};
    if (ok) z = *(const h16*)p;
    return z;
}

__global__ __launch_bounds__(256) void conv_wmma_kernel(const _Float16* __restrict__ xh,
                                                        const _Float16* __restrict__ weff,
                                                        const float* __restrict__ beff,
                                                        float* __restrict__ out) {
    const int b      = blockIdx.y;
    const int n_base = blockIdx.x * 64;
    const int tid  = threadIdx.x;
    const int wid  = tid >> 5;
    const int lane = tid & 31;
    const int lrow = lane & 15;
    const int hi   = lane >> 4;          // lane group 0/1

    // A (16-bit 16x32 layout): row = lane&15; halves = K[ko+16*hi .. +15]
    const _Float16* wrow = weff + ((size_t)(b * COUT + wid * 16 + lrow)) * KDIM;
    const int sel = hi * 16;
    // B (32x16 layout): col = lane&15; halves = K[ko+16*hi .. +15]
    const int khi = hi * 16;

    const _Float16* xb = xh + (size_t)b * HW_ * C_;

    int yt[4], xt[4];
    #pragma unroll
    for (int t = 0; t < 4; ++t) {
        const int n = n_base + t * 16 + lrow;
        yt[t] = n / W_;
        xt[t] = n % W_;
    }

    v8f acc0 = {}, acc1 = {}, acc2 = {}, acc3 = {};

    for (int p = 0; p < 9; ++p) {
        const int dy = p / 3 - 1, dx = p % 3 - 1;
        const _Float16* bp[4];
        bool ok[4];
        #pragma unroll
        for (int t = 0; t < 4; ++t) {
            const int iy = yt[t] + dy, ix = xt[t] + dx;
            ok[t] = (iy >= 0) & (iy < H_) & (ix >= 0) & (ix < W_);
            bp[t] = xb + (size_t)(iy * W_ + ix) * C_ + khi;
        }
        const _Float16* arow = wrow + p * 128 + sel;

        #pragma unroll
        for (int cq = 0; cq < 4; ++cq) {
            const int c0 = cq * 32;
            const h8 lo  = *(const h8*)(arow + c0);
            const h8 hiA = *(const h8*)(arow + c0 + 8);
            h16 a;
            #pragma unroll
            for (int j = 0; j < 8; ++j) { a[j] = lo[j]; a[8 + j] = hiA[j]; }

            const h16 b0 = load_b_frag(bp[0] + c0, ok[0]);
            const h16 b1 = load_b_frag(bp[1] + c0, ok[1]);
            const h16 b2 = load_b_frag(bp[2] + c0, ok[2]);
            const h16 b3 = load_b_frag(bp[3] + c0, ok[3]);

            acc0 = __builtin_amdgcn_wmma_f32_16x16x32_f16(false, a, false, b0, (short)0, acc0, false, false);
            acc1 = __builtin_amdgcn_wmma_f32_16x16x32_f16(false, a, false, b1, (short)0, acc1, false, false);
            acc2 = __builtin_amdgcn_wmma_f32_16x16x32_f16(false, a, false, b2, (short)0, acc2, false, false);
            acc3 = __builtin_amdgcn_wmma_f32_16x16x32_f16(false, a, false, b3, (short)0, acc3, false, false);
        }
    }

    // Epilogue: stage through padded LDS, then coalesced float4 NT stores.
    __shared__ float outs[128][65];      // 65: bank-conflict-free column writes
    const int mb = wid * 16 + hi * 8;    // C/D layout: VGPR r -> M = r + 8*hi
    #pragma unroll
    for (int r = 0; r < 8; ++r) {
        outs[mb + r][ 0 + lrow] = acc0[r];
        outs[mb + r][16 + lrow] = acc1[r];
        outs[mb + r][32 + lrow] = acc2[r];
        outs[mb + r][48 + lrow] = acc3[r];
    }
    __syncthreads();
    for (int idx = tid; idx < 128 * 16; idx += 256) {  // 2048 float4 chunks
        const int m   = idx >> 4;
        const int col = (idx & 15) * 4;
        const float bias = beff[b * COUT + m];
        vf4 v = { outs[m][col] + bias, outs[m][col + 1] + bias,
                  outs[m][col + 2] + bias, outs[m][col + 3] + bias };
        __builtin_nontemporal_store(v, (vf4*)(out + ((size_t)(b * COUT + m)) * HW_ + n_base + col));
    }
}

// ---------------------------------------------------------------------------
extern "C" void kernel_launch(void* const* d_in, const int* in_sizes, int n_in,
                              void* d_out, int out_size, void* d_ws, size_t ws_size,
                              hipStream_t stream) {
    const float* x    = (const float*)d_in[0];  // [32,128,56,56]
    const float* W    = (const float*)d_in[1];  // [4,128,128,3,3]
    const float* bvec = (const float*)d_in[2];  // [4,128]
    const float* aw1  = (const float*)d_in[3];  // [32,128]
    const float* aw2  = (const float*)d_in[4];  // [4,32]
    float* out = (float*)d_out;

    char* wsb = (char*)d_ws;
    float*    attn   = (float*)(wsb);                               //   512 B
    float*    pooled = (float*)(wsb + 512);                         // 16384 B
    float*    beff   = (float*)(wsb + 512 + 16384);                 // 16384 B
    _Float16* weff   = (_Float16*)(wsb + 33280);                    // 9437184 B
    _Float16* xh     = (_Float16*)(wsb + 33280 + (size_t)9437184);  // 25690112 B (NHWC)

    pool_kernel<<<dim3(C_, B_), 128, 0, stream>>>(x, pooled);
    attn_kernel<<<1, 32, 0, stream>>>(pooled, aw1, aw2, attn);
    weff_kernel<<<dim3(COUT, B_), 256, 0, stream>>>(W, bvec, attn, weff, beff);
    nhwc_kernel<<<dim3(HW_ / 64, C_ / 32, B_), 256, 0, stream>>>(x, xh);
    conv_wmma_kernel<<<dim3(HW_ / 64, B_), 256, 0, stream>>>(xh, weff, beff, out);
}
